// WaveNetModel_77060303224833
// MI455X (gfx1250) — matrix operand order repaired and checked by reading, not compiled, not run on hardware
//
#include <hip/hip_runtime.h>
#include <cmath>

// ---------------------------------------------------------------------------
// WaveNet on MI455X (gfx1250), wave32 + WMMA.
//
// Only the last OUT_LEN=256 timesteps feed the output; receptive field is
// 4*1023 = 4092, so we compute over a 4348-wide window. The 40 layers are
// strictly sequential, so k_layers uses ONE workgroup per batch element with
// the full hidden window resident in LDS (272 KB of the 320 KB WGP budget),
// iterating all 40 layers with only workgroup barriers. Descending-time tile
// sweep makes the in-place residual update race-free. All matmuls lower to
// v_wmma_f32_16x16x32_f16 (f16 operands, f32 accumulate). h0 staging uses
// the CDNA5 async global->LDS path (ASYNCcnt) when the builtin is available.
// ---------------------------------------------------------------------------

typedef _Float16 h16;
typedef __attribute__((ext_vector_type(16))) _Float16 v16h;
typedef __attribute__((ext_vector_type(8)))  _Float16 v8h;
typedef __attribute__((ext_vector_type(8)))  float    v8f;

#define NLAYERS 40
#define HC      32            // hidden channels
#define SC      256           // skip channels
#define CIN     256
#define LSEQ    8192
#define NBATCH  8
#define OUT_LEN 256
#define RF      4092          // 4 * (2^10 - 1)
#define WIN     (OUT_LEN + RF)   // 4348
#define PADL    8
#define NT      (WIN + PADL)     // 4356 valid time slots (j index)
#define NTA     4360             // padded allocation
#define T0ABS   (LSEQ - WIN - PADL) // abs time of j==0 -> 3836

__device__ __forceinline__ v8f wmma16(v16h a, v16h b, v8f c) {
  // D(16x16,f32) = A(16x32,f16) * B(32x16,f16) + C
  return __builtin_amdgcn_wmma_f32_16x16x32_f16(false, a, false, b,
                                                (short)0, c, false, false);
}

// Build an A-operand (16x32 f16) from f32 weights. Contiguous k order (same as the B loads):
// lanes 0-15 (M=lane) hold K = hi*16..hi*16+15, hi = lane>>4.
// p points at element (row, k=0) of this lane's row; s = stride between k.
__device__ __forceinline__ v16h loadA(const float* p, int s, int hi) {
  v16h a;
  const int k0 = hi * 16;
#pragma unroll
  for (int e = 0; e < 8; ++e) a[e]     = (h16)p[(k0 + e) * s];
#pragma unroll
  for (int e = 0; e < 8; ++e) a[8 + e] = (h16)p[(k0 + 8 + e) * s];
  return a;
}

__device__ __forceinline__ float fast_sigmoid(float x) {
  return 1.0f / (1.0f + __expf(-x));
}
__device__ __forceinline__ float fast_tanh(float x) {
#if __has_builtin(__builtin_amdgcn_tanhf)
  return __builtin_amdgcn_tanhf(x);   // v_tanh_f32 on gfx1250
#else
  return ::tanhf(x);
#endif
}

// ---------------------------------------------------------------------------
// Kernel A: start 1x1 conv (32 <- 256) over the pruned window only.
// h0[b][j][c] (f16, time-major rows of 32ch = 64B) for j in [0, NT).
// ---------------------------------------------------------------------------
__global__ __launch_bounds__(256)
void k_start(const float* __restrict__ x, const float* __restrict__ w0,
             h16* __restrict__ h0) {
  const int b    = blockIdx.y;
  const int wv   = threadIdx.x >> 5;
  const int lane = threadIdx.x & 31;
  const int n    = lane & 15, hi = lane >> 4;
  const int m    = wv & 1;                       // M tile (out-ch 16-block)
  const int j    = blockIdx.x * 64 + (wv >> 1) * 16 + n;
  const int t    = min(T0ABS + j, LSEQ - 1);     // clamp tail fringe
  const int oc0  = m * 16;
  const float* xb = x + (size_t)b * CIN * LSEQ;

  v8f acc = {};
#pragma unroll 1
  for (int ks = 0; ks < 8; ++ks) {               // K = 256 in 8 steps of 32
    const v16h a = loadA(w0 + (size_t)(oc0 + n) * CIN + ks * 32, 1, hi);
    v16h bv;
    const int ic0 = ks * 32 + hi * 16;           // B: lanes<16 K=0..15, else 16..31
#pragma unroll
    for (int e = 0; e < 16; ++e)
      bv[e] = (h16)xb[(size_t)(ic0 + e) * LSEQ + t];
    acc = wmma16(a, bv, acc);
  }
  if (j < NT) {
    v8h o;
#pragma unroll
    for (int v = 0; v < 8; ++v) o[v] = (h16)acc[v];
    *(v8h*)&h0[((size_t)b * NTA + j) * HC + oc0 + hi * 8] = o;
  }
}

// ---------------------------------------------------------------------------
// Kernel B: all 40 gated residual layers, one workgroup per batch element.
// Hidden window lives in LDS; skip accumulator lives in VGPRs.
// ---------------------------------------------------------------------------
__global__ __launch_bounds__(1024)
void k_layers(const h16* __restrict__ h0,
              const float* __restrict__ fws, const float* __restrict__ gws,
              const float* __restrict__ rws, const float* __restrict__ sws,
              float* __restrict__ skipbuf) {
  extern __shared__ h16 smem[];
  h16* hbuf = smem;                       // NTA * HC  (272.5 KB)
  h16* zbuf = smem + (size_t)NTA * HC;    // 256 * HC  (16 KB)

  const int b    = blockIdx.x;
  const int tid  = threadIdx.x;
  const int wv   = tid >> 5, lane = tid & 31;
  const int n    = lane & 15, hi = lane >> 4;
  const int m    = wv & 1;                // hidden M tile (0/1)
  const int ncol = wv >> 1;               // time 16-block within a 256 tile

  // Stage h0 window into LDS. Prefer the CDNA5 async global->LDS path
  // (global_load_async_to_lds_b128, tracked by ASYNCcnt): no VGPR round-trip.
  {
    const h16* src = h0 + (size_t)b * NTA * HC;
#if __has_builtin(__builtin_amdgcn_global_load_async_to_lds_b128)
    typedef int v4i_t __attribute__((ext_vector_type(4)));
    typedef __attribute__((address_space(3))) v4i_t lds_v4i;
    v4i_t*   gsrc = (v4i_t*)(uintptr_t)src;              // generic, non-const
    lds_v4i* ldst = (lds_v4i*)(unsigned)(uintptr_t)hbuf; // LDS offset bits
    const int total = NT * HC / 8;                       // 16B chunks
    for (int i = tid; i < total; i += 1024)
      __builtin_amdgcn_global_load_async_to_lds_b128(gsrc + i, ldst + i, 0, 0);
#if __has_builtin(__builtin_amdgcn_s_wait_asynccnt)
    __builtin_amdgcn_s_wait_asynccnt(0);
#else
    asm volatile("s_wait_asynccnt 0" ::: "memory");
#endif
#else
    const uint4* s4 = (const uint4*)src;
    uint4*       d4 = (uint4*)hbuf;
    const int total = NT * HC / 8;
    for (int i = tid; i < total; i += 1024) d4[i] = s4[i];
#endif
  }
  __syncthreads();

  v8f skipacc[8];
#pragma unroll
  for (int j8 = 0; j8 < 8; ++j8) { v8f z = {}; skipacc[j8] = z; }

  int alo = 0;                            // A_l: lowest valid index of h_l
  for (int l = 0; l < NLAYERS; ++l) {
    const int d      = 1 << (l % 10);
    const int awrite = alo + d;           // A_{l+1}
    const float* fw = fws + (size_t)l * HC * HC * 2;
    const float* gw = gws + (size_t)l * HC * HC * 2;
    const float* rw = rws + (size_t)l * HC * HC;
    const float* sw = sws + (size_t)l * SC * HC;

    const int oc = m * 16 + n;            // this lane's A row
    // Conv taps: k=0 -> x[t-d], k=1 -> x[t] (causal pad = d).
    const v16h af0 = loadA(fw + (size_t)oc * HC * 2 + 0, 2, hi);
    const v16h af1 = loadA(fw + (size_t)oc * HC * 2 + 1, 2, hi);
    const v16h ag0 = loadA(gw + (size_t)oc * HC * 2 + 0, 2, hi);
    const v16h ag1 = loadA(gw + (size_t)oc * HC * 2 + 1, 2, hi);
    const v16h ar  = loadA(rw + (size_t)oc * HC, 1, hi);

    const int ntile = (WIN - awrite + 255) >> 8;
    for (int tk = 0; tk < ntile; ++tk) {          // descending time sweep
      const int tstart = WIN - 256 * (tk + 1);    // >= -4 always
      const int i   = tstart + ncol * 16 + n;     // this lane's time column
      const int jt  = i + PADL;                   // >= 4, in-buffer
      const int jtd = max(i - d + PADL, 0);       // clamp: fringe reads garbage,
                                                  // results land below awrite
      // B operands: 32 contiguous bytes of a time-major LDS row.
      const v16h bt  = *(const v16h*)&hbuf[(size_t)jt  * HC + hi * 16];
      const v16h btd = *(const v16h*)&hbuf[(size_t)jtd * HC + hi * 16];

      v8f f = {}, g = {};
      f = wmma16(af0, btd, f); f = wmma16(af1, bt, f);
      g = wmma16(ag0, btd, g); g = wmma16(ag1, bt, g);

      v8h zv;
#pragma unroll
      for (int v = 0; v < 8; ++v)
        zv[v] = (h16)(fast_tanh(f[v]) * fast_sigmoid(g[v]));
      const int tl = ncol * 16 + n;
      *(v8h*)&zbuf[(size_t)tl * HC + m * 16 + hi * 8] = zv;
      __syncthreads();   // all h reads + z writes done before h writes / z reads

      const v16h bz = *(const v16h*)&zbuf[(size_t)tl * HC + hi * 16];
      v8f r = {};
      r = wmma16(ar, bz, r);
      {   // in-place residual: each lane owns its 8 contiguous (t, ch) halfs
        h16* hp = &hbuf[(size_t)jt * HC + m * 16 + hi * 8];
        v8h old = *(v8h*)hp, upd;
#pragma unroll
        for (int v = 0; v < 8; ++v) upd[v] = (h16)((float)old[v] + r[v]);
        *(v8h*)hp = upd;
      }
      if (tk == 0) {  // top tile == exactly the last-256 output window
#pragma unroll 1
        for (int j8 = 0; j8 < 8; ++j8) {
          const int sc = (m * 8 + j8) * 16 + n;
          const v16h as = loadA(sw + (size_t)sc * HC, 1, hi);
          skipacc[j8] = wmma16(as, bz, skipacc[j8]);
        }
      }
      __syncthreads();  // zbuf reuse hazard across tiles
    }
    alo = awrite;
  }

  // Spill VGPR-resident skip (256ch x 256t, f32) to scratch.
  float* sb = skipbuf + (size_t)b * SC * OUT_LEN;
  const int t = ncol * 16 + n;
#pragma unroll
  for (int j8 = 0; j8 < 8; ++j8)
#pragma unroll
    for (int v = 0; v < 8; ++v) {
      const int sc = (m * 8 + j8) * 16 + v + hi * 8;
      sb[(size_t)sc * OUT_LEN + t] = skipacc[j8][v];
    }
}

// ---------------------------------------------------------------------------
// Kernel C: out = end2 @ relu(end1 @ relu(skip)), transposed store.
// ---------------------------------------------------------------------------
__global__ __launch_bounds__(1024)
void k_end(const float* __restrict__ skipbuf, const float* __restrict__ e1,
           const float* __restrict__ e2, float* __restrict__ out) {
  extern __shared__ h16 smem[];
  h16* buf = smem;                         // 256 x 256 f16, time-major
  const int b    = blockIdx.x;
  const int tid  = threadIdx.x;
  const int wv   = tid >> 5, lane = tid & 31;
  const int n    = lane & 15, hi = lane >> 4;
  const int m    = wv & 1, ncol = wv >> 1;
  const int tcol = ncol * 16 + n;

  const float* sk = skipbuf + (size_t)b * SC * OUT_LEN;
  for (int idx = tid; idx < SC * OUT_LEN; idx += 1024) {
    const int c = idx >> 8, t = idx & 255;   // src layout [c][t]
    const float v = sk[idx];
    buf[(size_t)t * SC + c] = (h16)(v > 0.f ? v : 0.f);
  }
  __syncthreads();

  v8f acc[8];
#pragma unroll
  for (int j8 = 0; j8 < 8; ++j8) { v8f z = {}; acc[j8] = z; }
#pragma unroll 1
  for (int ks = 0; ks < 8; ++ks) {
    const v16h bv = *(const v16h*)&buf[(size_t)tcol * SC + ks * 32 + hi * 16];
#pragma unroll 1
    for (int j8 = 0; j8 < 8; ++j8) {
      const int oc = (m * 8 + j8) * 16 + n;
      const v16h a = loadA(e1 + (size_t)oc * SC + ks * 32, 1, hi);
      acc[j8] = wmma16(a, bv, acc[j8]);
    }
  }
  __syncthreads();           // all GEMM1 reads done before overwrite
#pragma unroll
  for (int j8 = 0; j8 < 8; ++j8) {
    const int oc0 = (m * 8 + j8) * 16 + hi * 8;
    v8h o;
#pragma unroll
    for (int v = 0; v < 8; ++v) {
      const float t2 = acc[j8][v];
      o[v] = (h16)(t2 > 0.f ? t2 : 0.f);
    }
    *(v8h*)&buf[(size_t)tcol * SC + oc0] = o;
  }
  __syncthreads();

#pragma unroll
  for (int j8 = 0; j8 < 8; ++j8) { v8f z = {}; acc[j8] = z; }
#pragma unroll 1
  for (int ks = 0; ks < 8; ++ks) {
    const v16h bv = *(const v16h*)&buf[(size_t)tcol * SC + ks * 32 + hi * 16];
#pragma unroll 1
    for (int j8 = 0; j8 < 8; ++j8) {
      const int oc = (m * 8 + j8) * 16 + n;
      const v16h a = loadA(e2 + (size_t)oc * SC + ks * 32, 1, hi);
      acc[j8] = wmma16(a, bv, acc[j8]);
    }
  }
  // Output layout: (b*256 + t)*256 + oc  (transpose handled here).
  float* ob = out + (size_t)b * OUT_LEN * SC;
#pragma unroll
  for (int j8 = 0; j8 < 8; ++j8) {
    const int oc0 = (m * 8 + j8) * 16 + hi * 8;
    v8f st;
#pragma unroll
    for (int v = 0; v < 8; ++v) st[v] = acc[j8][v];
    *(v8f*)&ob[(size_t)tcol * SC + oc0] = st;
  }
}

// ---------------------------------------------------------------------------
extern "C" void kernel_launch(void* const* d_in, const int* in_sizes, int n_in,
                              void* d_out, int out_size, void* d_ws, size_t ws_size,
                              hipStream_t stream) {
  const float* x   = (const float*)d_in[0];
  const float* w0  = (const float*)d_in[1];
  const float* fws = (const float*)d_in[2];
  const float* gws = (const float*)d_in[3];
  const float* rws = (const float*)d_in[4];
  const float* sws = (const float*)d_in[5];
  const float* e1  = (const float*)d_in[6];
  const float* e2  = (const float*)d_in[7];
  float* out = (float*)d_out;

  h16*   h0      = (h16*)d_ws;                                   // 2.23 MB
  float* skipbuf = (float*)((char*)d_ws +
                            (size_t)NBATCH * NTA * HC * sizeof(h16));  // 2 MB

  dim3 gA((NT + 63) / 64, NBATCH);
  k_start<<<gA, 256, 0, stream>>>(x, w0, h0);

  const size_t shB = ((size_t)NTA * HC + 256 * HC) * sizeof(h16);  // 295,424 B
  k_layers<<<NBATCH, 1024, shB, stream>>>(h0, fws, gws, rws, sws, skipbuf);

  const size_t shC = (size_t)SC * OUT_LEN * sizeof(h16);           // 131,072 B
  k_end<<<NBATCH, 1024, shC, stream>>>(skipbuf, e1, e2, out);
}
